// TransformerEncoderLayer_1090921693389
// MI455X (gfx1250) — matrix operand order repaired and checked by reading, not compiled, not run on hardware
//
#include <hip/hip_runtime.h>
#include <hip/hip_bf16.h>
#include <math.h>

#define SEQL   2048
#define DMODEL 1024
#define DINNER 2048
#define DFFN   4096
#define NHEAD  16
#define DHEAD  64
#define CHUNK  128
#define NCHUNK 16

typedef _Float16 half_t;
typedef __attribute__((ext_vector_type(16))) _Float16 v16h;
typedef __attribute__((ext_vector_type(8)))  float    v8f;

// ---------------------------------------------------------------------------
// WMMA helpers (gfx1250 wave32, V_WMMA_F32_16X16X32_F16)
// ---------------------------------------------------------------------------
__device__ __forceinline__ v8f wmma_f16(v16h a, v16h b, v8f c) {
  // 8 args: (neg_a, A, neg_b, B, c_mod, C, reuse_a, reuse_b)
  return __builtin_amdgcn_wmma_f32_16x16x32_f16(false, a, false, b, (short)0, c,
                                                false, false);
}

// A fragment: 16x32 f16 tile, row-major source, lda in elements.
// Lane L (wave32): m = L&15; halves h=0..15 -> K = (L<16?0:16)+h.
__device__ __forceinline__ v16h a_frag_f16(const half_t* A, int lda, int row0, int col0) {
  const int l = threadIdx.x & 31;
  const half_t* p = A + (size_t)(row0 + (l & 15)) * lda + col0;
  const int kb = (l < 16) ? 0 : 16;
  v16h r;
#pragma unroll
  for (int h = 0; h < 8; ++h) r[h] = p[kb + h];
#pragma unroll
  for (int h = 0; h < 8; ++h) r[8 + h] = p[kb + 8 + h];
  return r;
}

// A fragment read from LDS f32 (used for the P matrix in PV), converting to f16.
__device__ __forceinline__ v16h a_frag_lds_f32(const float* S, int lds, int col0) {
  const int l = threadIdx.x & 31;
  const float* p = S + (size_t)(l & 15) * lds + col0;
  const int kb = (l < 16) ? 0 : 16;
  v16h r;
#pragma unroll
  for (int h = 0; h < 8; ++h) r[h] = (_Float16)p[kb + h];
#pragma unroll
  for (int h = 0; h < 8; ++h) r[8 + h] = (_Float16)p[kb + 8 + h];
  return r;
}

// B fragment for C = A @ W^T where W is row-major N x K, f16.
// B_{k,n} = W[n][k].  Lane: n = n0 + (L&15); halves h -> k = k0 + (L<16?0:16) + h.
// 16 contiguous halves per lane -> two global_load_b128.
__device__ __forceinline__ v16h b_frag_wf16(const half_t* W, int ldw, int n0, int k0) {
  const int l = threadIdx.x & 31;
  const half_t* p = W + (size_t)(n0 + (l & 15)) * ldw + k0 + ((l < 16) ? 0 : 16);
  v16h r;
#pragma unroll
  for (int h = 0; h < 16; ++h) r[h] = p[h];
  return r;
}

// ---------------------------------------------------------------------------
// f32 -> f16 bulk convert (weights, once per launch; amortized over 128 m-tiles)
// ---------------------------------------------------------------------------
__global__ void cvt_h_kernel(const float* __restrict__ in, half_t* __restrict__ out,
                             size_t n) {
  const size_t i = (size_t)blockIdx.x * blockDim.x + threadIdx.x;
  if (i < n) out[i] = (half_t)in[i];
}

// V^T for attention PV: vt[d][key] = vh[key][d]  (DMODEL x SEQL, f16)
__global__ void vtrans_kernel(const half_t* __restrict__ vh, half_t* __restrict__ vt) {
  const size_t i = (size_t)blockIdx.x * blockDim.x + threadIdx.x;
  if (i >= (size_t)DMODEL * SEQL) return;
  const int d = (int)(i >> 11);          // / SEQL
  const int key = (int)(i & (SEQL - 1));
  vt[i] = vh[(size_t)key * DMODEL + d];
}

// ---------------------------------------------------------------------------
// Generic GEMM: out = act(A @ W^T + bias) [+ resid]
//   A: f16 MxK row-major, W: f16 NxK row-major.
//   Each wave computes a 16x64 output strip (4 WMMA n-subtiles, A reused 4x).
//   K-loop is software-pipelined with a BRANCH-FREE prefetch: the next chunk
//   offset is clamped to K-32 (last iteration redundantly re-loads in-bounds
//   data that is discarded), so the backend can track exact load counts and
//   overlap the 10 prefetch b128 loads with the current 4 WMMAs.
//   Out-of-range n-subtiles are clamped to N-16 (valid loads, results
//   discarded by the n<N epilogue guard).
// ---------------------------------------------------------------------------
__global__ __launch_bounds__(256) void gemm_aw_kernel(
    const half_t* __restrict__ A, const half_t* __restrict__ W,
    const float* __restrict__ bias, const float* __restrict__ resid,
    float* __restrict__ outF, half_t* __restrict__ outH,
    int M, int N, int K, int gelu) {
  const int gw = __builtin_amdgcn_readfirstlane(
      (int)((blockIdx.x * blockDim.x + threadIdx.x) >> 5));
  const int mtiles = M >> 4;
  const int nblks = (N + 63) >> 6;
  if (gw >= mtiles * nblks) return;   // wave-uniform exit
  const int row0 = (gw % mtiles) << 4;
  const int ncol0 = (gw / mtiles) << 6;
  // clamped sub-tile column bases (N is always a multiple of 16 here)
  const int n0 = ncol0;
  const int n1 = (ncol0 + 32 <= N) ? ncol0 + 16 : N - 16;
  const int n2 = (ncol0 + 48 <= N) ? ncol0 + 32 : N - 16;
  const int n3 = (ncol0 + 64 <= N) ? ncol0 + 48 : N - 16;

  v16h a_c  = a_frag_f16(A, K, row0, 0);
  v16h b0_c = b_frag_wf16(W, K, n0, 0);
  v16h b1_c = b_frag_wf16(W, K, n1, 0);
  v16h b2_c = b_frag_wf16(W, K, n2, 0);
  v16h b3_c = b_frag_wf16(W, K, n3, 0);

  v8f c0 = {}, c1 = {}, c2 = {}, c3 = {};
  for (int k0 = 0; k0 < K; k0 += 32) {
    // branch-free prefetch: clamp keeps the last iteration's loads in-bounds
    const int kn = (k0 + 32 < K) ? (k0 + 32) : k0;
    v16h a_n  = a_frag_f16(A, K, row0, kn);
    v16h b0_n = b_frag_wf16(W, K, n0, kn);
    v16h b1_n = b_frag_wf16(W, K, n1, kn);
    v16h b2_n = b_frag_wf16(W, K, n2, kn);
    v16h b3_n = b_frag_wf16(W, K, n3, kn);
    c0 = wmma_f16(a_c, b0_c, c0);
    c1 = wmma_f16(a_c, b1_c, c1);
    c2 = wmma_f16(a_c, b2_c, c2);
    c3 = wmma_f16(a_c, b3_c, c3);
    a_c = a_n; b0_c = b0_n; b1_c = b1_n; b2_c = b2_n; b3_c = b3_n;
  }

  const int l = threadIdx.x & 31;
  const int nlane = l & 15;
  const int rbase = row0 + ((l < 16) ? 0 : 8);
  v8f cs[4] = {c0, c1, c2, c3};
#pragma unroll
  for (int s = 0; s < 4; ++s) {
    const int n = ncol0 + s * 16 + nlane;
    if (n >= N) continue;                // clamped duplicates never pass this
    const float bv = bias ? bias[n] : 0.0f;
#pragma unroll
    for (int r = 0; r < 8; ++r) {
      const int row = rbase + r;
      float v = cs[s][r] + bv;
      if (gelu) {
        const float u = v;
        v = 0.5f * u * (1.0f + tanhf(0.7978845608f * (u + 0.044715f * u * u * u)));
      }
      if (resid) v += resid[(size_t)row * N + n];
      if (outF) outF[(size_t)row * N + n] = v;
      if (outH) outH[(size_t)row * N + n] = (half_t)v;
    }
  }
}

// ---------------------------------------------------------------------------
// LayerNorm over rows of length C (one block per row).
//   y = (x-mu)*rstd [*g+b] ; out = y [+ resid] ; written to f32 and/or f16.
// ---------------------------------------------------------------------------
__global__ __launch_bounds__(256) void ln_kernel(
    const float* __restrict__ x, const float* __restrict__ g,
    const float* __restrict__ b, const float* __restrict__ resid,
    float* __restrict__ outF, half_t* __restrict__ outH, int C) {
  __shared__ float red[256];
  const int row = blockIdx.x;
  const int tid = threadIdx.x;
  const float* xr = x + (size_t)row * C;

  float s = 0.0f;
  for (int i = tid; i < C; i += 256) s += xr[i];
  red[tid] = s; __syncthreads();
  for (int o = 128; o > 0; o >>= 1) { if (tid < o) red[tid] += red[tid + o]; __syncthreads(); }
  const float mu = red[0] / (float)C;
  __syncthreads();

  float v = 0.0f;
  for (int i = tid; i < C; i += 256) { const float d = xr[i] - mu; v += d * d; }
  red[tid] = v; __syncthreads();
  for (int o = 128; o > 0; o >>= 1) { if (tid < o) red[tid] += red[tid + o]; __syncthreads(); }
  const float rstd = rsqrtf(red[0] / (float)C + 1e-5f);

  for (int i = tid; i < C; i += 256) {
    float y = (xr[i] - mu) * rstd;
    if (g) y = y * g[i] + b[i];
    if (resid) y += resid[(size_t)row * C + i];
    if (outF) outF[(size_t)row * C + i] = y;
    if (outH) outH[(size_t)row * C + i] = (half_t)y;
  }
}

// LN over 16-wide slices: raw is SEQL x 32 (cols 64..95 of xp proj); slice 0 -> Bm,
// slice 1 -> Cm.  (The 96-wide LN in the reference is affine per row, so the inner
// 16-wide LNs are invariant to it — exact simplification.)
__global__ void ln16_kernel(const float* __restrict__ raw, float* __restrict__ Bm,
                            float* __restrict__ Cm) {
  const int t = blockIdx.x * blockDim.x + threadIdx.x;
  if (t >= 2 * SEQL) return;
  const int row = t >> 1, sl = t & 1;
  const float* p = raw + (size_t)row * 32 + sl * 16;
  float mu = 0.0f;
#pragma unroll
  for (int i = 0; i < 16; ++i) mu += p[i];
  mu *= (1.0f / 16.0f);
  float var = 0.0f;
#pragma unroll
  for (int i = 0; i < 16; ++i) { const float d = p[i] - mu; var += d * d; }
  const float rstd = rsqrtf(var * (1.0f / 16.0f) + 1e-5f);
  float* o = (sl ? Cm : Bm) + (size_t)row * 16;
#pragma unroll
  for (int i = 0; i < 16; ++i) o[i] = (p[i] - mu) * rstd;
}

// ---------------------------------------------------------------------------
// Attention: one workgroup per (head, 16-query-row block).
//  Phase 1: S[16][2048] = (q K^T)/8 via WMMA into LDS.
//  Phase 2: exact softmax per row, in place (wave-per-2-rows, shfl reductions).
//  Phase 3: O = P @ V^T-layout via WMMA, K split over 8 waves, deterministic sum.
// Dynamic LDS: 16*2048 (S) + 8*16*64 (per-wave O partials) floats = 160 KB.
// ---------------------------------------------------------------------------
__global__ __launch_bounds__(256) void attn_kernel(
    const half_t* __restrict__ qh, const half_t* __restrict__ kh,
    const half_t* __restrict__ vt, half_t* __restrict__ oh) {
  extern __shared__ float smem[];
  float* S = smem;                    // 16 * SEQL
  float* Opart = smem + 16 * SEQL;    // 8 * 16 * 64
  const int head = blockIdx.x & (NHEAD - 1);
  const int qb = blockIdx.x >> 4;
  const int wave = __builtin_amdgcn_readfirstlane((int)(threadIdx.x >> 5));
  const int l = threadIdx.x & 31;
  const int col0 = head * DHEAD;

  // q fragments for this 16-row block (d-chunks of 32)
  v16h aq0 = a_frag_f16(qh, DMODEL, qb * 16, col0);
  v16h aq1 = a_frag_f16(qh, DMODEL, qb * 16, col0 + 32);

  // Phase 1: scores
  for (int j = wave; j < SEQL / 16; j += 8) {
    v8f c = {};
    c = wmma_f16(aq0, b_frag_wf16(kh, DMODEL, j * 16, col0), c);
    c = wmma_f16(aq1, b_frag_wf16(kh, DMODEL, j * 16, col0 + 32), c);
    const int rbase = (l < 16) ? 0 : 8;
#pragma unroll
    for (int r = 0; r < 8; ++r)
      S[(rbase + r) * SEQL + j * 16 + (l & 15)] = c[r] * 0.125f;
  }
  __syncthreads();

  // Phase 2: softmax (wave `wave` handles rows 2*wave, 2*wave+1)
#pragma unroll
  for (int rr = 0; rr < 2; ++rr) {
    float* Sr = S + (wave * 2 + rr) * SEQL;
    float mx = -1e30f;
    for (int i = l; i < SEQL; i += 32) mx = fmaxf(mx, Sr[i]);
#pragma unroll
    for (int o = 16; o > 0; o >>= 1) mx = fmaxf(mx, __shfl_xor(mx, o));
    float sum = 0.0f;
    for (int i = l; i < SEQL; i += 32) { const float e = expf(Sr[i] - mx); Sr[i] = e; sum += e; }
#pragma unroll
    for (int o = 16; o > 0; o >>= 1) sum += __shfl_xor(sum, o);
    const float inv = 1.0f / sum;
    for (int i = l; i < SEQL; i += 32) Sr[i] *= inv;
  }
  __syncthreads();

  // Phase 3: O = P @ V ; wave w covers keys [w*256, w*256+256).
  // vt is DMODEL x SEQL (V transposed), so B-frags are contiguous 32B loads.
  v8f c0 = {}, c1 = {}, c2 = {}, c3 = {};
  for (int kc = 0; kc < 8; ++kc) {
    const int kbase = wave * 256 + kc * 32;
    v16h a = a_frag_lds_f32(S, SEQL, kbase);
    c0 = wmma_f16(a, b_frag_wf16(vt, SEQL, col0, kbase), c0);
    c1 = wmma_f16(a, b_frag_wf16(vt, SEQL, col0 + 16, kbase), c1);
    c2 = wmma_f16(a, b_frag_wf16(vt, SEQL, col0 + 32, kbase), c2);
    c3 = wmma_f16(a, b_frag_wf16(vt, SEQL, col0 + 48, kbase), c3);
  }
  {
    const int rbase = (l < 16) ? 0 : 8;
    const int nlane = l & 15;
    float* Ow = Opart + wave * (16 * 64);
    v8f cc[4] = {c0, c1, c2, c3};
#pragma unroll
    for (int s = 0; s < 4; ++s)
#pragma unroll
      for (int r = 0; r < 8; ++r)
        Ow[(rbase + r) * 64 + s * 16 + nlane] = cc[s][r];
  }
  __syncthreads();

  // Deterministic reduction over the 8 wave partials + store
  for (int i = threadIdx.x; i < 16 * 64; i += 256) {
    float acc = 0.0f;
#pragma unroll
    for (int w = 0; w < 8; ++w) acc += Opart[w * (16 * 64) + i];
    const int row = i >> 6, col = i & 63;
    oh[(size_t)(qb * 16 + row) * DMODEL + col0 + col] = (half_t)acc;
  }
}

// ---------------------------------------------------------------------------
// Mamba: depthwise causal conv (D_CONV=4) over time, output f16 for xp GEMM.
// ---------------------------------------------------------------------------
__global__ void conv_kernel(const float* __restrict__ x3, const float* __restrict__ cw,
                            const float* __restrict__ cb, half_t* __restrict__ xch) {
  const size_t idx = (size_t)blockIdx.x * blockDim.x + threadIdx.x;
  if (idx >= (size_t)SEQL * DINNER) return;
  const int ch = (int)(idx & (DINNER - 1));
  const int t = (int)(idx >> 11);
  float acc = cb[ch];
#pragma unroll
  for (int j = 0; j < 4; ++j) {
    const int tt = t - 3 + j;
    if (tt >= 0) acc += cw[ch * 4 + j] * x3[(size_t)tt * DINNER + ch];
  }
  xch[idx] = (half_t)acc;
}

// ---------------------------------------------------------------------------
// Selective scan with constant M (delta cancels in the reference's max-shift):
//   M_ij = clip(exp(min(logA) - logA_ij), 1e-4, 1e4), logA_ij = log|A[j][i]|+1e-8
// ---------------------------------------------------------------------------
__global__ __launch_bounds__(256) void scan_m_kernel(const float* __restrict__ A,
                                                     float* __restrict__ Mb) {
  __shared__ float red[256];
  const int e = threadIdx.x;
  const int i = e >> 4, j = e & 15;
  const float la = logf(fabsf(A[j * 16 + i]) + 1e-8f);
  red[e] = la; __syncthreads();
  for (int o = 128; o > 0; o >>= 1) { if (e < o) red[e] = fminf(red[e], red[e + o]); __syncthreads(); }
  float m = expf(red[0] - la);
  m = fminf(fmaxf(m, 1e-4f), 1e4f);
  Mb[e] = m;
}

// Powers M^1..M^128 (Pb[(k-1)*256 + i*16 + j])
__global__ __launch_bounds__(256) void scan_pow_kernel(const float* __restrict__ Mb,
                                                       float* __restrict__ Pb) {
  __shared__ float M[256], cur[256];
  const int e = threadIdx.x;
  const int i = e >> 4, j = e & 15;
  M[e] = Mb[e]; cur[e] = Mb[e]; Pb[e] = Mb[e];
  __syncthreads();
  for (int k = 2; k <= CHUNK; ++k) {
    float acc = 0.0f;
#pragma unroll
    for (int t = 0; t < 16; ++t) acc += M[i * 16 + t] * cur[t * 16 + j];
    __syncthreads();
    cur[e] = acc; Pb[(k - 1) * 256 + e] = acc;
    __syncthreads();
  }
}

// Phase 1: local scans L_t (zero carry) per 128-step chunk; one wave per chunk.
__global__ __launch_bounds__(32) void scan_local_kernel(
    const float* __restrict__ Bm, const float* __restrict__ Mb,
    float* __restrict__ L, float* __restrict__ Lend) {
  const int c = blockIdx.x;
  const int l = threadIdx.x;
  const int i = l & 15;
  const bool act = l < 16;
  float Mrow[16];
#pragma unroll
  for (int j = 0; j < 16; ++j) Mrow[j] = Mb[i * 16 + j];
  const int t0 = c * CHUNK;
  float h = act ? Bm[(size_t)t0 * 16 + i] : 0.0f;
  if (act) L[(size_t)t0 * 16 + i] = h;
  for (int t = t0 + 1; t < t0 + CHUNK; ++t) {
    float hn = act ? Bm[(size_t)t * 16 + i] : 0.0f;
#pragma unroll
    for (int j = 0; j < 16; ++j) hn += Mrow[j] * __shfl(h, j);
    h = hn;
    if (act) L[(size_t)t * 16 + i] = h;
  }
  if (act) Lend[c * 16 + i] = h;
}

// Phase 2: carry recurrence E_c = Lend[c] + M^128 E_{c-1}; carry[c] = E_{c-1}.
__global__ __launch_bounds__(32) void scan_carry_kernel(
    const float* __restrict__ Lend, const float* __restrict__ Pb,
    float* __restrict__ carry) {
  const int l = threadIdx.x;
  const int i = l & 15;
  const bool act = l < 16;
  float P[16];
#pragma unroll
  for (int j = 0; j < 16; ++j) P[j] = Pb[(CHUNK - 1) * 256 + i * 16 + j];
  float h = 0.0f;
  for (int c = 0; c < NCHUNK; ++c) {
    if (act) carry[c * 16 + i] = h;
    float hn = act ? Lend[c * 16 + i] : 0.0f;
#pragma unroll
    for (int j = 0; j < 16; ++j) hn += P[j] * __shfl(h, j);
    h = hn;
  }
}

// Phase 3: H_t = L_t + M^{(t-t0)+1} * carry[chunk]; s[t] = <H_t, C_t>.
__global__ __launch_bounds__(128) void scan_final_kernel(
    const float* __restrict__ L, const float* __restrict__ Pb,
    const float* __restrict__ carry, const float* __restrict__ Cm,
    float* __restrict__ s) {
  const int c = blockIdx.x;
  const int ti = threadIdx.x;
  const int t = c * CHUNK + ti;
  float cr[16];
#pragma unroll
  for (int j = 0; j < 16; ++j) cr[j] = carry[c * 16 + j];
  const float* P = Pb + (size_t)ti * 256;  // M^(ti+1)
  float dot = 0.0f;
#pragma unroll
  for (int i = 0; i < 16; ++i) {
    float hi = L[(size_t)t * 16 + i];
#pragma unroll
    for (int j = 0; j < 16; ++j) hi += P[i * 16 + j] * cr[j];
    dot += hi * Cm[(size_t)t * 16 + i];
  }
  s[t] = dot;
}

// y = clip(x3*(s[t] + Dp[ch]), +/-1000) -> f16 for the out-proj GEMM
__global__ void ymul_kernel(const float* __restrict__ x3, const float* __restrict__ s,
                            const float* __restrict__ Dp, half_t* __restrict__ yh) {
  const size_t idx = (size_t)blockIdx.x * blockDim.x + threadIdx.x;
  if (idx >= (size_t)SEQL * DINNER) return;
  const int ch = (int)(idx & (DINNER - 1));
  const int t = (int)(idx >> 11);
  float v = x3[idx] * (s[t] + Dp[ch]);
  v = fminf(fmaxf(v, -1000.0f), 1000.0f);
  yh[idx] = (half_t)v;
}

// ---------------------------------------------------------------------------
extern "C" void kernel_launch(void* const* d_in, const int* in_sizes, int n_in,
                              void* d_out, int out_size, void* d_ws, size_t ws_size,
                              hipStream_t stream) {
  const float* x    = (const float*)d_in[0];
  const float* Wq   = (const float*)d_in[1];
  const float* bq   = (const float*)d_in[2];
  const float* Wk   = (const float*)d_in[3];
  const float* bk   = (const float*)d_in[4];
  const float* Wv   = (const float*)d_in[5];
  const float* bv   = (const float*)d_in[6];
  const float* Wo   = (const float*)d_in[7];
  const float* bo   = (const float*)d_in[8];
  const float* fc1w = (const float*)d_in[9];
  const float* fc1b = (const float*)d_in[10];
  const float* fc2w = (const float*)d_in[11];
  const float* fc2b = (const float*)d_in[12];
  const float* inw  = (const float*)d_in[13];
  const float* inb  = (const float*)d_in[14];
  const float* cw   = (const float*)d_in[15];
  const float* cb   = (const float*)d_in[16];
  const float* xpw  = (const float*)d_in[17];
  const float* xpb  = (const float*)d_in[18];
  const float* Amat = (const float*)d_in[19];
  const float* Dp   = (const float*)d_in[20];
  /* d_in[21] = dt_p: provably dead (delta cancels in lt - lt.max()) */
  const float* outw = (const float*)d_in[22];
  const float* outb = (const float*)d_in[23];
  const float* g1   = (const float*)d_in[24];
  const float* be1  = (const float*)d_in[25];
  const float* g2   = (const float*)d_in[26];
  const float* be2  = (const float*)d_in[27];
  const float* g3   = (const float*)d_in[28];
  const float* be3  = (const float*)d_in[29];
  float* out = (float*)d_out;
  (void)in_sizes; (void)n_in; (void)out_size; (void)ws_size;

  // ---- scratch layout (bump allocator) ----
  char* wp = (char*)d_ws;
  auto alloc = [&](size_t bytes) -> char* {
    char* p = wp;
    wp += (bytes + 255) & ~(size_t)255;
    return p;
  };
  // f16 weight copies (converted once per launch)
  half_t* wqh  = (half_t*)alloc((size_t)DMODEL * DMODEL * 2);
  half_t* wkh  = (half_t*)alloc((size_t)DMODEL * DMODEL * 2);
  half_t* wvh  = (half_t*)alloc((size_t)DMODEL * DMODEL * 2);
  half_t* woh  = (half_t*)alloc((size_t)DMODEL * DMODEL * 2);
  half_t* inwh = (half_t*)alloc((size_t)DINNER * DMODEL * 2);
  half_t* xpwh = (half_t*)alloc((size_t)32 * DINNER * 2);     // rows 64..95 only
  half_t* outwh= (half_t*)alloc((size_t)DMODEL * DINNER * 2);
  half_t* fc1h = (half_t*)alloc((size_t)DFFN * DMODEL * 2);
  half_t* fc2h = (half_t*)alloc((size_t)DMODEL * DFFN * 2);
  // activations
  half_t* h1h  = (half_t*)alloc((size_t)SEQL * DMODEL * 2);
  half_t* qh   = (half_t*)alloc((size_t)SEQL * DMODEL * 2);
  half_t* kh   = (half_t*)alloc((size_t)SEQL * DMODEL * 2);
  half_t* vh   = (half_t*)alloc((size_t)SEQL * DMODEL * 2);
  half_t* vt   = (half_t*)alloc((size_t)DMODEL * SEQL * 2);
  half_t* ohb  = (half_t*)alloc((size_t)SEQL * DMODEL * 2);
  float*  x2   = (float*)alloc((size_t)SEQL * DMODEL * 4);
  float*  t1   = (float*)alloc((size_t)SEQL * DMODEL * 4);
  half_t* h2h  = (half_t*)alloc((size_t)SEQL * DMODEL * 2);
  float*  x3   = (float*)alloc((size_t)SEQL * DINNER * 4);
  half_t* xch  = (half_t*)alloc((size_t)SEQL * DINNER * 2);
  float*  bcr  = (float*)alloc((size_t)SEQL * 32 * 4);
  float*  Bm   = (float*)alloc((size_t)SEQL * 16 * 4);
  float*  Cm   = (float*)alloc((size_t)SEQL * 16 * 4);
  float*  Mb   = (float*)alloc(256 * 4);
  float*  Pb   = (float*)alloc((size_t)CHUNK * 256 * 4);
  float*  Lb   = (float*)alloc((size_t)SEQL * 16 * 4);
  float*  Lend = (float*)alloc(NCHUNK * 16 * 4);
  float*  carry= (float*)alloc(NCHUNK * 16 * 4);
  float*  sbuf = (float*)alloc((size_t)SEQL * 4);
  half_t* yh   = (half_t*)alloc((size_t)SEQL * DINNER * 2);
  float*  mam  = (float*)alloc((size_t)SEQL * DMODEL * 4);
  float*  x4   = (float*)alloc((size_t)SEQL * DMODEL * 4);
  half_t* h3h  = (half_t*)alloc((size_t)SEQL * DMODEL * 2);
  half_t* ff1h = (half_t*)alloc((size_t)SEQL * DFFN * 2);

  auto cvt = [&](const float* src, half_t* dst, size_t n) {
    cvt_h_kernel<<<(unsigned)((n + 255) / 256), 256, 0, stream>>>(src, dst, n);
  };
  auto gemm = [&](const half_t* A, const half_t* W, const float* bias,
                  const float* resid, float* oF, half_t* oH, int M, int N, int K,
                  int gelu) {
    const int tiles = (M / 16) * ((N + 63) / 64);
    const int blocks = (tiles + 7) / 8;
    gemm_aw_kernel<<<blocks, 256, 0, stream>>>(A, W, bias, resid, oF, oH, M, N, K, gelu);
  };

  // ---- weight conversion (amortized over all 128 m-tiles of each GEMM) ----
  cvt(Wq, wqh, (size_t)DMODEL * DMODEL);
  cvt(Wk, wkh, (size_t)DMODEL * DMODEL);
  cvt(Wv, wvh, (size_t)DMODEL * DMODEL);
  cvt(Wo, woh, (size_t)DMODEL * DMODEL);
  cvt(inw, inwh, (size_t)DINNER * DMODEL);
  cvt(xpw + (size_t)64 * DINNER, xpwh, (size_t)32 * DINNER);
  cvt(outw, outwh, (size_t)DMODEL * DINNER);
  cvt(fc1w, fc1h, (size_t)DFFN * DMODEL);
  cvt(fc2w, fc2h, (size_t)DMODEL * DFFN);

  // ---- attention branch ----
  ln_kernel<<<SEQL, 256, 0, stream>>>(x, g1, be1, nullptr, nullptr, h1h, DMODEL);
  gemm(h1h, wqh, bq, nullptr, nullptr, qh, SEQL, DMODEL, DMODEL, 0);
  gemm(h1h, wkh, bk, nullptr, nullptr, kh, SEQL, DMODEL, DMODEL, 0);
  gemm(h1h, wvh, bv, nullptr, nullptr, vh, SEQL, DMODEL, DMODEL, 0);
  vtrans_kernel<<<(DMODEL * SEQL) / 256, 256, 0, stream>>>(vh, vt);

  const int attn_lds = (16 * SEQL + 8 * 16 * 64) * (int)sizeof(float);  // 160 KB
  (void)hipFuncSetAttribute((const void*)attn_kernel,
                            hipFuncAttributeMaxDynamicSharedMemorySize, attn_lds);
  attn_kernel<<<(SEQL / 16) * NHEAD, 256, attn_lds, stream>>>(qh, kh, vt, ohb);

  gemm(ohb, woh, bo, x, x2, nullptr, SEQL, DMODEL, DMODEL, 0);  // x2 = x + attn

  // ---- mamba branch ----
  ln_kernel<<<SEQL, 256, 0, stream>>>(x2, g2, be2, nullptr, t1, nullptr, DMODEL);
  ln_kernel<<<SEQL, 256, 0, stream>>>(t1, nullptr, nullptr, nullptr, nullptr, h2h, DMODEL);
  gemm(h2h, inwh, inb, nullptr, x3, nullptr, SEQL, DINNER, DMODEL, 0);
  conv_kernel<<<(SEQL * DINNER) / 256, 256, 0, stream>>>(x3, cw, cb, xch);
  // only cols 64..95 of the xp projection survive (delta dead, 96-LN affine)
  gemm(xch, xpwh, xpb + 64, nullptr, bcr, nullptr, SEQL, 32, DINNER, 0);
  ln16_kernel<<<(2 * SEQL) / 256, 256, 0, stream>>>(bcr, Bm, Cm);

  scan_m_kernel<<<1, 256, 0, stream>>>(Amat, Mb);
  scan_pow_kernel<<<1, 256, 0, stream>>>(Mb, Pb);
  scan_local_kernel<<<NCHUNK, 32, 0, stream>>>(Bm, Mb, Lb, Lend);
  scan_carry_kernel<<<1, 32, 0, stream>>>(Lend, Pb, carry);
  scan_final_kernel<<<NCHUNK, CHUNK, 0, stream>>>(Lb, Pb, carry, Cm, sbuf);

  ymul_kernel<<<(SEQL * DINNER) / 256, 256, 0, stream>>>(x3, sbuf, Dp, yh);
  gemm(yh, outwh, outb, nullptr, mam, nullptr, SEQL, DMODEL, DINNER, 0);
  ln_kernel<<<SEQL, 256, 0, stream>>>(mam, nullptr, nullptr, x2, x4, nullptr, DMODEL);

  // ---- FFN branch ----
  ln_kernel<<<SEQL, 256, 0, stream>>>(x4, g3, be3, nullptr, nullptr, h3h, DMODEL);
  gemm(h3h, fc1h, fc1b, nullptr, nullptr, ff1h, SEQL, DFFN, DMODEL, 1);   // GELU
  gemm(ff1h, fc2h, fc2b, x4, out, nullptr, SEQL, DMODEL, DFFN, 0);        // + resid
}
